// Encoder_26036091749017
// MI455X (gfx1250) — compile-verified, matrix operand order checked by reading, not run on hardware
//
#include <hip/hip_runtime.h>
#include <math.h>

// ---------------------------------------------------------------------------
// Problem constants (from the reference)
// ---------------------------------------------------------------------------
#define N_NODES 80000
#define N_EDGES 480000
#define NPAD    80128              // N + 128 (one spare 128-row block), a multiple of 128
#define HID     128
#define KPAD    160                // padded K for (HID+DIM)=131 -> 5 WMMA k-steps
#define EF      (N_EDGES + N_NODES) // edges + self loops

typedef __attribute__((ext_vector_type(16))) _Float16 v16h;
typedef __attribute__((ext_vector_type(8)))  _Float16 v8h;
typedef __attribute__((ext_vector_type(8)))  float    v8f;

union AFrag { v16h v; v8h h[2]; };

// order-preserving float<->uint key for atomicMax-based segment max
__device__ __forceinline__ unsigned fkey(float f) {
  unsigned u = __float_as_uint(f);
  return (u & 0x80000000u) ? ~u : (u | 0x80000000u);
}
__device__ __forceinline__ float fdec(unsigned k) {
  return __uint_as_float((k & 0x80000000u) ? (k ^ 0x80000000u) : ~k);
}

// ---------------------------------------------------------------------------
// Utility kernels
// ---------------------------------------------------------------------------
__global__ void k_zero(float* p, long n) {
  long i = (long)blockIdx.x * 256 + threadIdx.x;
  if (i < n) p[i] = 0.f;
}

// pack [x | pos | y] -> fp16, K=11 padded into first 32 cols of A[NPAD][KPAD]
__global__ void k_pack0(const float* __restrict__ x, const float* __restrict__ pos,
                        const float* __restrict__ y, _Float16* __restrict__ A) {
  long i = (long)blockIdx.x * 256 + threadIdx.x;
  if (i >= (long)NPAD * 32) return;
  long row = i >> 5; int col = (int)(i & 31);
  float v = 0.f;
  if (row < N_NODES) {
    if (col < 5)       v = x[row * 5 + col];
    else if (col < 8)  v = pos[row * 3 + (col - 5)];
    else if (col < 11) v = y[0];
  }
  A[row * KPAD + col] = (_Float16)v;
}

// pack [h | pos] -> fp16 A[NPAD][KPAD], K=131, zero-padded
__global__ void k_packH(const float* __restrict__ h, const float* __restrict__ pos,
                        _Float16* __restrict__ A) {
  long i = (long)blockIdx.x * 256 + threadIdx.x;
  if (i >= (long)NPAD * KPAD) return;
  long row = i / KPAD; int col = (int)(i % KPAD);
  float v = 0.f;
  if (row < N_NODES) {
    if (col < HID)          v = h[row * HID + col];
    else if (col < HID + 3) v = pos[row * 3 + (col - HID)];
  }
  A[i] = (_Float16)v;
}

// transpose-pack weights W[K][128] (fp32 row-major) -> Wt[128][KPAD] fp16
__global__ void k_packW(const float* __restrict__ W, int K, _Float16* __restrict__ Wt) {
  int i = blockIdx.x * 256 + threadIdx.x;
  if (i >= 128 * KPAD) return;
  int n = i / KPAD, k = i % KPAD;
  Wt[i] = (_Float16)((k < K) ? W[k * 128 + n] : 0.f);
}

// ---------------------------------------------------------------------------
// WMMA GEMM: Out[M][128] = A_fp16[*][KPAD] @ Wt_fp16^T  (+ bias)
// 256 threads = 8 waves; block covers 128 rows; wave covers a 16x128 strip.
// Weights staged in LDS (128*KPAD fp16 = 40 KB), reused by all 8 waves.
// Fragment addressing follows the CDNA5 16-bit A/B VGPR layout:
//   lane<16 holds K = {kb+0..7, kb+16..23}; lane>=16 holds {kb+8..15, kb+24..31}.
// sched_group_barrier pins the per-k-step schedule to
//   DSread x6  ->  (WMMA x1 -> DSread x2) x5  ->  WMMA x3
// so B-fragment ds_loads run 3 fragments ahead of their consuming WMMA and
// the allocator must keep distinct rotating buffers (graduated s_wait_dscnt
// instead of a full drain per WMMA).
// ---------------------------------------------------------------------------
__global__ __launch_bounds__(256)
void k_gemm(const _Float16* __restrict__ A, const _Float16* __restrict__ Wt,
            const float* __restrict__ bias, float* __restrict__ Out,
            int M, int ksteps) {
  extern __shared__ _Float16 sW[];
  for (int i = threadIdx.x * 8; i < 128 * KPAD; i += 256 * 8)
    *(v8h*)(sW + i) = *(const v8h*)(Wt + i);
  __syncthreads();

  int lane = threadIdx.x & 31, wave = threadIdx.x >> 5;
  int l16 = lane & 15, hi = lane >> 4;
  long rowBase = (long)blockIdx.x * 128 + wave * 16;

  v8f acc[8] = {};
  const _Float16* aRow = A + (rowBase + l16) * KPAD;
  const _Float16* bRow = sW + (size_t)l16 * KPAD + hi * 8;

  for (int ks = 0; ks < ksteps; ks++) {
    int kb = ks * 32;
    AFrag af;
    af.h[0] = *(const v8h*)(aRow + kb + hi * 8);
    af.h[1] = *(const v8h*)(aRow + kb + 16 + hi * 8);
    if (ks + 1 < ksteps) __builtin_prefetch(aRow + kb + 32, 0, 1); // global_prefetch_b8

    // 4-deep rotating B-fragment pipeline
    AFrag bb[4];
#pragma unroll
    for (int t = 0; t < 3; t++) {
      const _Float16* bp = bRow + (size_t)t * 16 * KPAD + kb;
      bb[t].h[0] = *(const v8h*)bp;
      bb[t].h[1] = *(const v8h*)(bp + 16);
    }
#pragma unroll
    for (int t = 0; t < 8; t++) {
      if (t + 3 < 8) {
        const _Float16* bp = bRow + (size_t)(t + 3) * 16 * KPAD + kb;
        bb[(t + 3) & 3].h[0] = *(const v8h*)bp;
        bb[(t + 3) & 3].h[1] = *(const v8h*)(bp + 16);
      }
      acc[t] = __builtin_amdgcn_wmma_f32_16x16x32_f16(
          false, af.v, false, bb[t & 3].v, (short)0, acc[t], false, false);
    }
#if __has_builtin(__builtin_amdgcn_sched_group_barrier)
    // Pin the schedule: 6 DS reads ahead, then interleave WMMA with DS reads.
    __builtin_amdgcn_sched_group_barrier(0x100, 6, 0);   // DS read x6
#pragma unroll
    for (int i5 = 0; i5 < 5; ++i5) {
      __builtin_amdgcn_sched_group_barrier(0x008, 1, 0); // WMMA x1
      __builtin_amdgcn_sched_group_barrier(0x100, 2, 0); // DS read x2
    }
    __builtin_amdgcn_sched_group_barrier(0x008, 3, 0);   // WMMA x3 (drain)
#endif
  }
  // C layout: VGPR r, lanes 0-15: M=r,N=lane ; lanes 16-31: M=r+8,N=lane-16
#pragma unroll
  for (int t = 0; t < 8; t++)
#pragma unroll
    for (int r = 0; r < 8; r++) {
      long row = rowBase + r + hi * 8;
      if (row < M) {
        int col = t * 16 + l16;
        float v = acc[t][r];
        if (bias) v += bias[col];
        Out[row * 128 + col] = v;
      }
    }
}

// ---------------------------------------------------------------------------
// Graph prep: in-degree + mean incoming edge attr (self-loop fill value)
// ---------------------------------------------------------------------------
__global__ void k_edge_prep(const int* __restrict__ ei, const float* __restrict__ pos,
                            float* deg, float* lsum) {
  int e = blockIdx.x * 256 + threadIdx.x;
  if (e >= N_EDGES) return;
  int s = ei[e], d = ei[N_EDGES + e];
  atomicAdd(&deg[d], 1.f);
#pragma unroll
  for (int i = 0; i < 3; i++)
    atomicAdd(&lsum[d * 3 + i], pos[d * 3 + i] - pos[s * 3 + i]);
}
__global__ void k_loop_div(float* lsum, const float* deg) {
  int n = blockIdx.x * 256 + threadIdx.x;
  if (n >= N_NODES) return;
  float dg = fmaxf(deg[n], 1.f);
#pragma unroll
  for (int i = 0; i < 3; i++) lsum[n * 3 + i] /= dg;
}

// ---------------------------------------------------------------------------
// GATv2 edge phase (128 channels): one wave32 per edge, 4 channels/lane.
// ---------------------------------------------------------------------------
__global__ __launch_bounds__(256)
void k_edge_logit(const int* __restrict__ ei, const float* __restrict__ pos,
                  const float* __restrict__ lattr,
                  const float* __restrict__ xl, const float* __restrict__ xr,
                  const float* __restrict__ We, const float* __restrict__ att,
                  float* __restrict__ logit, unsigned* __restrict__ mxkey) {
  int f = blockIdx.x * 8 + (threadIdx.x >> 5);
  if (f >= EF) return;
  int lane = threadIdx.x & 31;
  int s, d; float e0, e1, e2;
  if (f < N_EDGES) {
    s = ei[f]; d = ei[N_EDGES + f];
    e0 = pos[d*3+0] - pos[s*3+0];
    e1 = pos[d*3+1] - pos[s*3+1];
    e2 = pos[d*3+2] - pos[s*3+2];
  } else {
    int n = f - N_EDGES; s = d = n;
    e0 = lattr[n*3+0]; e1 = lattr[n*3+1]; e2 = lattr[n*3+2];
  }
  int c = lane * 4;
  float4 xlv = *(const float4*)(xl + (long)s * 128 + c);
  float4 xrv = *(const float4*)(xr + (long)d * 128 + c);
  float4 w0  = *(const float4*)(We + c);
  float4 w1  = *(const float4*)(We + 128 + c);
  float4 w2  = *(const float4*)(We + 256 + c);
  float4 av  = *(const float4*)(att + c);
  float r = 0.f, m;
  m = xlv.x + xrv.x + e0*w0.x + e1*w1.x + e2*w2.x; m = m > 0.f ? m : 0.2f*m; r += m*av.x;
  m = xlv.y + xrv.y + e0*w0.y + e1*w1.y + e2*w2.y; m = m > 0.f ? m : 0.2f*m; r += m*av.y;
  m = xlv.z + xrv.z + e0*w0.z + e1*w1.z + e2*w2.z; m = m > 0.f ? m : 0.2f*m; r += m*av.z;
  m = xlv.w + xrv.w + e0*w0.w + e1*w1.w + e2*w2.w; m = m > 0.f ? m : 0.2f*m; r += m*av.w;
  for (int o = 16; o; o >>= 1) r += __shfl_xor(r, o, 32);
  if (lane == 0) {
    logit[f] = r;
    atomicMax(&mxkey[d], fkey(r));
  }
}

__global__ void k_edge_expsum(const int* __restrict__ ei, const float* __restrict__ logit,
                              const unsigned* __restrict__ mxkey,
                              float* __restrict__ zbuf, float* __restrict__ denom) {
  int f = blockIdx.x * 256 + threadIdx.x;
  if (f >= EF) return;
  int d = (f < N_EDGES) ? ei[N_EDGES + f] : (f - N_EDGES);
  float z = __expf(logit[f] - fdec(mxkey[d]));
  zbuf[f] = z;
  atomicAdd(&denom[d], z);
}

__global__ __launch_bounds__(256)
void k_edge_agg(const int* __restrict__ ei, const float* __restrict__ zbuf,
                const float* __restrict__ denom, const float* __restrict__ xl,
                float* __restrict__ hnext) {
  int f = blockIdx.x * 8 + (threadIdx.x >> 5);
  if (f >= EF) return;
  int lane = threadIdx.x & 31;
  int s, d;
  if (f < N_EDGES) { s = ei[f]; d = ei[N_EDGES + f]; }
  else             { s = d = f - N_EDGES; }
  float alpha = zbuf[f] / denom[d];
  int c = lane * 4;
  float4 v = *(const float4*)(xl + (long)s * 128 + c);
  atomicAdd(&hnext[(long)d * 128 + c + 0], alpha * v.x);
  atomicAdd(&hnext[(long)d * 128 + c + 1], alpha * v.y);
  atomicAdd(&hnext[(long)d * 128 + c + 2], alpha * v.z);
  atomicAdd(&hnext[(long)d * 128 + c + 3], alpha * v.w);
}

__global__ void k_finish_elu(const float* __restrict__ hnext, const float* __restrict__ b,
                             float* __restrict__ h) {
  long i = (long)blockIdx.x * 256 + threadIdx.x;
  if (i >= (long)N_NODES * 128) return;
  float v = hnext[i] + b[i & 127];
  h[i] = v > 0.f ? v : (__expf(v) - 1.f);
}

// ---------------------------------------------------------------------------
// SAGPool score GNN (C=1): scalar per-node projections + scalar edge softmax
// ---------------------------------------------------------------------------
__global__ __launch_bounds__(256)
void k_pool_lin(const _Float16* __restrict__ A, const float* __restrict__ pWl,
                const float* __restrict__ pWr, float* xl1, float* xr1) {
  int n = blockIdx.x * 8 + (threadIdx.x >> 5);
  if (n >= N_NODES) return;
  int lane = threadIdx.x & 31;
  float al = 0.f, ar = 0.f;
  for (int k = lane; k < 131; k += 32) {
    float a = (float)A[(long)n * KPAD + k];
    al += a * pWl[k]; ar += a * pWr[k];
  }
  for (int o = 16; o; o >>= 1) { al += __shfl_xor(al, o, 32); ar += __shfl_xor(ar, o, 32); }
  if (lane == 0) { xl1[n] = al; xr1[n] = ar; }
}

__global__ void k_pool_logit(const int* __restrict__ ei, const float* __restrict__ pos,
                             const float* __restrict__ lattr,
                             const float* __restrict__ xl1, const float* __restrict__ xr1,
                             const float* __restrict__ pWe, const float* __restrict__ pa,
                             float* __restrict__ logit, unsigned* __restrict__ mxkey) {
  int f = blockIdx.x * 256 + threadIdx.x;
  if (f >= EF) return;
  int s, d; float e0, e1, e2;
  if (f < N_EDGES) {
    s = ei[f]; d = ei[N_EDGES + f];
    e0 = pos[d*3+0]-pos[s*3+0]; e1 = pos[d*3+1]-pos[s*3+1]; e2 = pos[d*3+2]-pos[s*3+2];
  } else {
    int n = f - N_EDGES; s = d = n;
    e0 = lattr[n*3+0]; e1 = lattr[n*3+1]; e2 = lattr[n*3+2];
  }
  float m = xl1[s] + xr1[d] + e0*pWe[0] + e1*pWe[1] + e2*pWe[2];
  m = m > 0.f ? m : 0.2f * m;
  float r = m * pa[0];
  logit[f] = r;
  atomicMax(&mxkey[d], fkey(r));
}

__global__ void k_pool_agg(const int* __restrict__ ei, const float* __restrict__ zbuf,
                           const float* __restrict__ denom, const float* __restrict__ xl1,
                           float* __restrict__ score) {
  int f = blockIdx.x * 256 + threadIdx.x;
  if (f >= EF) return;
  int s, d;
  if (f < N_EDGES) { s = ei[f]; d = ei[N_EDGES + f]; }
  else             { s = d = f - N_EDGES; }
  atomicAdd(&score[d], (zbuf[f] / denom[d]) * xl1[s]);
}

__global__ void k_pool_finish(float* score, const float* pb) {
  int n = blockIdx.x * 256 + threadIdx.x;
  if (n >= N_NODES) return;
  score[n] = tanhf(score[n] + pb[0]);
}

__global__ void k_gate(float* h, const float* __restrict__ score) {
  long i = (long)blockIdx.x * 256 + threadIdx.x;
  if (i >= (long)N_NODES * 128) return;
  h[i] *= score[i >> 7];
}

__global__ void k_out(const float* __restrict__ src, float* __restrict__ dst,
                      long n_src, long n_out) {
  long i = (long)blockIdx.x * 256 + threadIdx.x;
  if (i >= n_out) return;
  dst[i] = (i < n_src) ? src[i] : 0.f;
}

// ---------------------------------------------------------------------------
// Host driver
// ---------------------------------------------------------------------------
extern "C" void kernel_launch(void* const* d_in, const int* in_sizes, int n_in,
                              void* d_out, int out_size, void* d_ws, size_t ws_size,
                              hipStream_t stream) {
  const float* x    = (const float*)d_in[0];
  const float* y    = (const float*)d_in[1];
  const float* pos  = (const float*)d_in[2];
  const int*   ei   = (const int*)  d_in[3];
  const float* Wl0  = (const float*)d_in[4];
  const float* Wr0  = (const float*)d_in[5];
  const float* We0  = (const float*)d_in[6];
  const float* a0   = (const float*)d_in[7];
  const float* b0   = (const float*)d_in[8];
  const float* Wls  = (const float*)d_in[9];
  const float* Wrs  = (const float*)d_in[10];
  const float* Wes  = (const float*)d_in[11];
  const float* atts = (const float*)d_in[12];
  const float* bs   = (const float*)d_in[13];
  const float* pWl  = (const float*)d_in[14];
  const float* pWr  = (const float*)d_in[15];
  const float* pWe  = (const float*)d_in[16];
  const float* pa   = (const float*)d_in[17];
  const float* pb   = (const float*)d_in[18];
  const float* linW = (const float*)d_in[19];
  const float* linb = (const float*)d_in[20];
  (void)in_sizes; (void)n_in;

  // workspace carve-up
  size_t off = 0;
  auto alloc = [&](size_t bytes) { size_t o = off; off = (off + bytes + 255) & ~size_t(255); return o; };
  char* ws = (char*)d_ws;
  size_t oA   = alloc((size_t)NPAD * KPAD * 2);
  size_t oWtL = alloc((size_t)128 * KPAD * 2);
  size_t oWtR = alloc((size_t)128 * KPAD * 2);
  size_t oXl  = alloc((size_t)NPAD * 128 * 4);
  size_t oXr  = alloc((size_t)NPAD * 128 * 4);
  size_t oH   = alloc((size_t)N_NODES * 128 * 4);
  size_t oHn  = alloc((size_t)N_NODES * 128 * 4);
  size_t oDeg = alloc((size_t)N_NODES * 4);
  size_t oLat = alloc((size_t)N_NODES * 3 * 4);
  size_t oMx  = alloc((size_t)N_NODES * 4);
  size_t oDen = alloc((size_t)N_NODES * 4);
  size_t oLog = alloc((size_t)EF * 4);
  size_t oZ   = alloc((size_t)EF * 4);
  size_t oXl1 = alloc((size_t)N_NODES * 4);
  size_t oXr1 = alloc((size_t)N_NODES * 4);
  size_t oSc  = alloc((size_t)N_NODES * 4);
  if (off > ws_size) return;  // not enough scratch — refuse to run OOB

  _Float16* A   = (_Float16*)(ws + oA);
  _Float16* WtL = (_Float16*)(ws + oWtL);
  _Float16* WtR = (_Float16*)(ws + oWtR);
  float* xl  = (float*)(ws + oXl);
  float* xr  = (float*)(ws + oXr);
  float* h   = (float*)(ws + oH);
  float* hn  = (float*)(ws + oHn);
  float* deg = (float*)(ws + oDeg);
  float* lat = (float*)(ws + oLat);
  unsigned* mx = (unsigned*)(ws + oMx);
  float* den = (float*)(ws + oDen);
  float* lg  = (float*)(ws + oLog);
  float* zb  = (float*)(ws + oZ);
  float* xl1 = (float*)(ws + oXl1);
  float* xr1 = (float*)(ws + oXr1);
  float* sc  = (float*)(ws + oSc);

  const int T = 256;
  auto blocks = [](long n) { return (int)((n + 255) / 256); };
  const size_t shW = (size_t)128 * KPAD * 2;   // 40 KB weight stage in LDS

  auto run_conv = [&](const float* Wl, const float* Wr, int K, int ksteps,
                      const float* We, const float* att, const float* b) {
    k_packW<<<blocks(128L * KPAD), T, 0, stream>>>(Wl, K, WtL);
    k_packW<<<blocks(128L * KPAD), T, 0, stream>>>(Wr, K, WtR);
    k_gemm<<<NPAD / 128, T, shW, stream>>>(A, WtL, nullptr, xl, N_NODES, ksteps);
    k_gemm<<<NPAD / 128, T, shW, stream>>>(A, WtR, nullptr, xr, N_NODES, ksteps);
    k_zero<<<blocks(N_NODES), T, 0, stream>>>((float*)mx, N_NODES);
    k_zero<<<blocks(N_NODES), T, 0, stream>>>(den, N_NODES);
    k_zero<<<blocks((long)N_NODES * 128), T, 0, stream>>>(hn, (long)N_NODES * 128);
    k_edge_logit<<<blocks((long)EF * 32), T, 0, stream>>>(ei, pos, lat, xl, xr, We, att, lg, mx);
    k_edge_expsum<<<blocks(EF), T, 0, stream>>>(ei, lg, mx, zb, den);
    k_edge_agg<<<blocks((long)EF * 32), T, 0, stream>>>(ei, zb, den, xl, hn);
    k_finish_elu<<<blocks((long)N_NODES * 128), T, 0, stream>>>(hn, b, h);
  };

  auto run_pool = [&](int l) {
    k_packH<<<blocks((long)NPAD * KPAD), T, 0, stream>>>(h, pos, A);
    k_pool_lin<<<blocks((long)N_NODES * 32), T, 0, stream>>>(A, pWl + l * 131, pWr + l * 131, xl1, xr1);
    k_zero<<<blocks(N_NODES), T, 0, stream>>>((float*)mx, N_NODES);
    k_zero<<<blocks(N_NODES), T, 0, stream>>>(den, N_NODES);
    k_zero<<<blocks(N_NODES), T, 0, stream>>>(sc, N_NODES);
    k_pool_logit<<<blocks(EF), T, 0, stream>>>(ei, pos, lat, xl1, xr1, pWe + l * 3, pa + l, lg, mx);
    k_edge_expsum<<<blocks(EF), T, 0, stream>>>(ei, lg, mx, zb, den);
    k_pool_agg<<<blocks(EF), T, 0, stream>>>(ei, zb, den, xl1, sc);
    k_pool_finish<<<blocks(N_NODES), T, 0, stream>>>(sc, pb + l);
    k_gate<<<blocks((long)N_NODES * 128), T, 0, stream>>>(h, sc);
  };

  // one-time graph prep: degree + mean incoming edge-attr (self-loop fill)
  k_zero<<<blocks(N_NODES), T, 0, stream>>>(deg, N_NODES);
  k_zero<<<blocks((long)N_NODES * 3), T, 0, stream>>>(lat, (long)N_NODES * 3);
  k_edge_prep<<<blocks(N_EDGES), T, 0, stream>>>(ei, pos, deg, lat);
  k_loop_div<<<blocks(N_NODES), T, 0, stream>>>(lat, deg);

  // conv0: [x|pos|y] (K=11 -> 1 k-step)
  k_pack0<<<blocks((long)NPAD * 32), T, 0, stream>>>(x, pos, y, A);
  run_conv(Wl0, Wr0, 11, 1, We0, a0, b0);

  // conv1 (Wls[0]): [h|pos] (K=131 -> 5 k-steps)
  k_packH<<<blocks((long)NPAD * KPAD), T, 0, stream>>>(h, pos, A);
  run_conv(Wls + 0 * 131 * 128, Wrs + 0 * 131 * 128, 131, 5,
           Wes + 0 * 3 * 128, atts + 0 * 128, bs + 0 * 128);

  // pool 0 + conv2 (Wls[1])
  run_pool(0);
  k_packH<<<blocks((long)NPAD * KPAD), T, 0, stream>>>(h, pos, A);
  run_conv(Wls + 1 * 131 * 128, Wrs + 1 * 131 * 128, 131, 5,
           Wes + 1 * 3 * 128, atts + 1 * 128, bs + 1 * 128);

  // pool 1 + conv3 (Wls[2])
  run_pool(1);
  k_packH<<<blocks((long)NPAD * KPAD), T, 0, stream>>>(h, pos, A);
  run_conv(Wls + 2 * 131 * 128, Wrs + 2 * 131 * 128, 131, 5,
           Wes + 2 * 3 * 128, atts + 2 * 128, bs + 2 * 128);

  // last conv (Wls[3])
  k_packH<<<blocks((long)NPAD * KPAD), T, 0, stream>>>(h, pos, A);
  run_conv(Wls + 3 * 131 * 128, Wrs + 3 * 131 * 128, 131, 5,
           Wes + 3 * 3 * 128, atts + 3 * 128, bs + 3 * 128);

  // final linear: out = h @ linW + linb  (K=128 -> 4 k-steps; pos cols unread)
  k_packH<<<blocks((long)NPAD * KPAD), T, 0, stream>>>(h, pos, A);
  k_packW<<<blocks(128L * KPAD), T, 0, stream>>>(linW, 128, WtL);
  k_gemm<<<NPAD / 128, T, shW, stream>>>(A, WtL, linb, xr, N_NODES, 4);

  k_out<<<blocks(out_size), T, 0, stream>>>(xr, (float*)d_out,
                                            (long)N_NODES * 128, (long)out_size);
}
